// LSTM_old_26852135535256
// MI455X (gfx1250) — matrix operand order repaired and checked by reading, not compiled, not run on hardware
//
#include <hip/hip_runtime.h>
#include <math.h>

typedef __attribute__((ext_vector_type(16))) _Float16 v16h;
typedef __attribute__((ext_vector_type(8)))  float    v8f;
typedef __attribute__((ext_vector_type(4)))  float    v4f;

#define WMMA(A, Bf, C) __builtin_amdgcn_wmma_f32_16x16x32_f16( \
    false, (A), false, (Bf), (short)0, (C), false, false)

// B-fragment fetch from LDS: fragment 'fi' (pre-swizzled), this lane's 16 halves (32B).
#define BF(base, fi) (*(const v16h*)((base) + (((fi) * 32 + lane) << 4)))

// Pack 8 consecutive f32 (two float4s) into fragment slots [base, base+8).
#define PACK8(frag, base, lo, hi)                                         \
    frag[(base) + 0] = (_Float16)lo[0]; frag[(base) + 1] = (_Float16)lo[1]; \
    frag[(base) + 2] = (_Float16)lo[2]; frag[(base) + 3] = (_Float16)lo[3]; \
    frag[(base) + 4] = (_Float16)hi[0]; frag[(base) + 5] = (_Float16)hi[1]; \
    frag[(base) + 6] = (_Float16)hi[2]; frag[(base) + 7] = (_Float16)hi[3];

__device__ __forceinline__ float sigm_(float x) {
    return 1.0f / (1.0f + __expf(-x));
}
__device__ __forceinline__ float tanh_(float x) {
    float e2 = __expf(2.0f * x);
    return (e2 - 1.0f) / (e2 + 1.0f);
}

__global__ __launch_bounds__(256) void lstm_step_wmma(
    const float* __restrict__ x,   const float* __restrict__ h0,
    const float* __restrict__ c0,
    const float* __restrict__ Wih, const float* __restrict__ Whh,
    const float* __restrict__ bih, const float* __restrict__ bhh,
    const float* __restrict__ Wo,  const float* __restrict__ bo,
    float* __restrict__ out, int Bsz)
{
    // f16 weights staged in LDS, pre-swizzled into WMMA B-fragment layout:
    // B (32x16 f16): lane = h*16 + n ; element j -> K = h*8 + (j&7) + (j>>3)*16, N = n.
    __shared__ __align__(32) _Float16 sWih[16384]; // 16 N-tiles x 2 k-steps
    __shared__ __align__(32) _Float16 sWhh[16384]; // 16 N-tiles x 2 k-steps
    __shared__ __align__(32) _Float16 sWo [ 8192]; //  4 N-tiles x 4 k-steps

    const int tid = threadIdx.x;

    for (int idx = tid; idx < 16384; idx += 256) {
        int j  = idx & 15;
        int ln = (idx >> 4) & 31;
        int fr = idx >> 9;              // nt*2 + ks
        int nt = fr >> 1, ks = fr & 1;
        int N  = nt * 16 + (ln & 15);
        int K  = ks * 32 + (ln >> 4) * 8 + (j & 7) + (j >> 3) * 16;
        sWih[idx] = (_Float16)Wih[N * 64 + K];
        sWhh[idx] = (_Float16)Whh[N * 64 + K];
    }
    for (int idx = tid; idx < 8192; idx += 256) {
        int j  = idx & 15;
        int ln = (idx >> 4) & 31;
        int fr = idx >> 9;              // nt*4 + ks
        int nt = fr >> 2, ks = fr & 3;
        int N  = nt * 16 + (ln & 15);
        int K  = ks * 32 + (ln >> 4) * 8 + (j & 7) + (j >> 3) * 16;
        sWo[idx] = (_Float16)Wo[N * 128 + K];
    }
    __syncthreads();

    const int lane  = tid & 31;
    const int wave  = tid >> 5;
    const int cn    = lane & 15;   // N (column) within 16-wide tile
    const int hh    = lane >> 4;   // wave-half selector
    const int tiles = Bsz >> 4;

    float* outLP = out;
    float* outH  = out + (size_t)Bsz * 64;
    float* outC  = out + (size_t)Bsz * 128;

    for (int rt = blockIdx.x * 8 + wave; rt < tiles; rt += gridDim.x * 8) {
        // Launder a zero offset each iteration so the compiler cannot hoist the
        // (loop-invariant) LDS B-fragment reads out of the loop (640 VGPRs of
        // fragments would spill to scratch). In-loop ds_load_b128 is cheap.
        unsigned zoff = 0;
        asm volatile("" : "+v"(zoff));
        const _Float16* bWih = sWih + zoff;
        const _Float16* bWhh = sWhh + zoff;
        const _Float16* bWo  = sWo  + zoff;

        // ---- A fragments (16x32 f16 per frag): lane holds row M=cn,
        //      K = hh*8 + j (j<8) and 16 + hh*8 + (j-8). Two frags cover K=0..63.
        // Batch all 16 b128 loads into distinct temps first (load clause),
        // then convert — overlaps HBM latency instead of load/wait/load/wait.
        const int arow = rt * 16 + cn;
        const float* xr = x  + (size_t)arow * 64 + hh * 8;
        const float* hr = h0 + (size_t)arow * 64 + hh * 8;
        v4f tx[8], th[8];
        #pragma unroll
        for (int q = 0; q < 4; ++q) {
            tx[2 * q]     = *(const v4f*)(xr + q * 16);
            tx[2 * q + 1] = *(const v4f*)(xr + q * 16 + 4);
            th[2 * q]     = *(const v4f*)(hr + q * 16);
            th[2 * q + 1] = *(const v4f*)(hr + q * 16 + 4);
        }
        v16h ax0, ax1, ah0, ah1;
        PACK8(ax0, 0, tx[0], tx[1]); PACK8(ax0, 8, tx[2], tx[3]);
        PACK8(ax1, 0, tx[4], tx[5]); PACK8(ax1, 8, tx[6], tx[7]);
        PACK8(ah0, 0, th[0], th[1]); PACK8(ah0, 8, th[2], th[3]);
        PACK8(ah1, 0, th[4], th[5]); PACK8(ah1, 8, th[6], th[7]);

        // ---- LSTM gates: 4 groups of 16 hidden columns. Gate tiles for group
        // hc are (i,f,g,o) = (hc, hc+4, hc+8, hc+12); same lane slot everywhere.
        // Accumulators start at inline 0 (WMMA SRC2 = 0); bias folded into the
        // epilogue to avoid 8-wide splats feeding the matrix pipe.
        #pragma unroll
        for (int hc = 0; hc < 4; ++hc) {
            const int ci = hc * 16 + cn;
            const float b_i = bih[ci]       + bhh[ci];
            const float b_f = bih[ci + 64]  + bhh[ci + 64];
            const float b_g = bih[ci + 128] + bhh[ci + 128];
            const float b_o = bih[ci + 192] + bhh[ci + 192];
            v8f ai = {}, af = {}, ag = {}, ao = {};

            ai = WMMA(ax0, BF(bWih, (hc     ) * 2 + 0), ai);
            ai = WMMA(ax1, BF(bWih, (hc     ) * 2 + 1), ai);
            ai = WMMA(ah0, BF(bWhh, (hc     ) * 2 + 0), ai);
            ai = WMMA(ah1, BF(bWhh, (hc     ) * 2 + 1), ai);

            af = WMMA(ax0, BF(bWih, (hc +  4) * 2 + 0), af);
            af = WMMA(ax1, BF(bWih, (hc +  4) * 2 + 1), af);
            af = WMMA(ah0, BF(bWhh, (hc +  4) * 2 + 0), af);
            af = WMMA(ah1, BF(bWhh, (hc +  4) * 2 + 1), af);

            ag = WMMA(ax0, BF(bWih, (hc +  8) * 2 + 0), ag);
            ag = WMMA(ax1, BF(bWih, (hc +  8) * 2 + 1), ag);
            ag = WMMA(ah0, BF(bWhh, (hc +  8) * 2 + 0), ag);
            ag = WMMA(ah1, BF(bWhh, (hc +  8) * 2 + 1), ag);

            ao = WMMA(ax0, BF(bWih, (hc + 12) * 2 + 0), ao);
            ao = WMMA(ax1, BF(bWih, (hc + 12) * 2 + 1), ao);
            ao = WMMA(ah0, BF(bWhh, (hc + 12) * 2 + 0), ao);
            ao = WMMA(ah1, BF(bWhh, (hc + 12) * 2 + 1), ao);

            // Elementwise LSTM update; C/D layout: VGPR r -> row r + 8*hh.
            #pragma unroll
            for (int r = 0; r < 8; ++r) {
                const int m = rt * 16 + r + hh * 8;
                const size_t off = (size_t)m * 64 + hc * 16 + cn;
                float ig = sigm_(ai[r] + b_i);
                float fg = sigm_(af[r] + b_f);
                float gg = tanh_(ag[r] + b_g);
                float og = sigm_(ao[r] + b_o);
                float cnew = fg * c0[off] + ig * gg;
                outH[off] = og * tanh_(cnew);
                outC[off] = cnew;
            }
        }

        // ---- Output head: [x,h] @ W_o^T, then log-softmax over 64 cols.
        v8f L0 = {}, L1 = {}, L2 = {}, L3 = {};
        const float b0 = bo[cn],      b1 = bo[16 + cn];
        const float b2 = bo[32 + cn], b3 = bo[48 + cn];
        L0 = WMMA(ax0, BF(bWo, 0 * 4 + 0), L0);
        L0 = WMMA(ax1, BF(bWo, 0 * 4 + 1), L0);
        L0 = WMMA(ah0, BF(bWo, 0 * 4 + 2), L0);
        L0 = WMMA(ah1, BF(bWo, 0 * 4 + 3), L0);
        L1 = WMMA(ax0, BF(bWo, 1 * 4 + 0), L1);
        L1 = WMMA(ax1, BF(bWo, 1 * 4 + 1), L1);
        L1 = WMMA(ah0, BF(bWo, 1 * 4 + 2), L1);
        L1 = WMMA(ah1, BF(bWo, 1 * 4 + 3), L1);
        L2 = WMMA(ax0, BF(bWo, 2 * 4 + 0), L2);
        L2 = WMMA(ax1, BF(bWo, 2 * 4 + 1), L2);
        L2 = WMMA(ah0, BF(bWo, 2 * 4 + 2), L2);
        L2 = WMMA(ah1, BF(bWo, 2 * 4 + 3), L2);
        L3 = WMMA(ax0, BF(bWo, 3 * 4 + 0), L3);
        L3 = WMMA(ax1, BF(bWo, 3 * 4 + 1), L3);
        L3 = WMMA(ah0, BF(bWo, 3 * 4 + 2), L3);
        L3 = WMMA(ah1, BF(bWo, 3 * 4 + 3), L3);

        #pragma unroll
        for (int r = 0; r < 8; ++r) {
            const float l0 = L0[r] + b0, l1 = L1[r] + b1;
            const float l2 = L2[r] + b2, l3 = L3[r] + b3;
            // Row reduction over 64 cols: 4 local tiles + 16-lane half butterfly.
            float mx = fmaxf(fmaxf(l0, l1), fmaxf(l2, l3));
            #pragma unroll
            for (int d = 1; d < 16; d <<= 1)
                mx = fmaxf(mx, __shfl_xor(mx, d, 32));
            float s = __expf(l0 - mx) + __expf(l1 - mx) +
                      __expf(l2 - mx) + __expf(l3 - mx);
            #pragma unroll
            for (int d = 1; d < 16; d <<= 1)
                s += __shfl_xor(s, d, 32);
            const float lse = mx + __logf(s);
            const int m = rt * 16 + r + hh * 8;
            const size_t base = (size_t)m * 64 + cn;
            outLP[base]      = l0 - lse;
            outLP[base + 16] = l1 - lse;
            outLP[base + 32] = l2 - lse;
            outLP[base + 48] = l3 - lse;
        }
    }
}

extern "C" void kernel_launch(void* const* d_in, const int* in_sizes, int n_in,
                              void* d_out, int out_size, void* d_ws, size_t ws_size,
                              hipStream_t stream) {
    const float* x   = (const float*)d_in[0];
    const float* h0  = (const float*)d_in[1];
    const float* c0  = (const float*)d_in[2];
    const float* Wih = (const float*)d_in[3];
    const float* Whh = (const float*)d_in[4];
    const float* bih = (const float*)d_in[5];
    const float* bhh = (const float*)d_in[6];
    const float* Wo  = (const float*)d_in[7];
    const float* bo  = (const float*)d_in[8];
    float* out = (float*)d_out;

    const int Bsz = in_sizes[0] / 64;   // 524288
    dim3 grid(512), block(256);         // 8 waves/WG * 16 rows/wave, 8 sweeps
    lstm_step_wmma<<<grid, block, 0, stream>>>(
        x, h0, c0, Wih, Whh, bih, bhh, Wo, bo, out, Bsz);
}
